// MultiHeadAttention_2877628088499
// MI455X (gfx1250) — matrix operand order repaired and checked by reading, not compiled, not run on hardware
//
#include <hip/hip_runtime.h>
#include <hip/hip_bf16.h>

// ---------------------------------------------------------------------------
// MI455X (gfx1250) multi-head causal attention, bf16 WMMA everywhere.
//   x:[2,2048,1024] f32, w_*:[1024,1024] f32 (y = x @ W^T), out f32 [2,2048,1024]
// Pipeline: cvt(f32->bf16) -> 3x GEMM (Q,K,Vt) -> flash attention -> GEMM (Wo)
// GEMM staging per 64-deep k-stage (double buffered in LDS):
//   A tile (128x64 bf16, 16KB) : TDM tensor_load_to_lds (TENSORcnt), with
//       hardware LDS padding 16B per 128B row -> 144B pitch (36 banks)
//   W tile (128x64 bf16, 16KB) : per-lane GLOBAL_LOAD_ASYNC_TO_LDS_B128
//       (ASYNCcnt) into the same padded pitch
// Fragments then come from LDS via ds_load_b128.
// ---------------------------------------------------------------------------

typedef __attribute__((ext_vector_type(16))) __bf16 v16bf;
typedef __attribute__((ext_vector_type(8)))  __bf16 v8bf;
typedef __attribute__((ext_vector_type(4)))  __bf16 v4bf;
typedef __attribute__((ext_vector_type(8)))  float  v8f;
typedef __attribute__((ext_vector_type(4)))  unsigned int u32x4;
typedef __attribute__((ext_vector_type(8)))  int          i32x8;

#define N_EMBD   1024
#define N_HEAD   16
#define HEAD_DIM 64
#define BATCH    2
#define SEQ      2048
#define MROWS    (BATCH * SEQ)   // 4096

__device__ __forceinline__ v16bf cat8(v8bf a, v8bf b) {
  v16bf r;
#pragma unroll
  for (int i = 0; i < 8; ++i) { r[i] = a[i]; r[i + 8] = b[i]; }
  return r;
}

__device__ __forceinline__ v8f wmma_bf16(v16bf a, v16bf b, v8f c) {
  // D = A(16x32 bf16) * B(32x16 bf16) + C(16x16 f32)
  return __builtin_amdgcn_wmma_f32_16x16x32_bf16(
      /*neg_a=*/false, a, /*neg_b=*/false, b,
      /*c_mod=*/(short)0, c, /*reuse_a=*/false, /*reuse_b=*/false);
}

// Async global->LDS 16B copy (per-lane). LDS offset = low 32 bits of the
// generic __shared__ pointer (flat-LDS aperture keeps the wave-relative
// offset in addr[31:0]); tracked by ASYNCcnt, drained with s_wait_asynccnt.
__device__ __forceinline__ void async_ld_b128(unsigned lds_byte_off,
                                              const void* gaddr) {
  asm volatile("global_load_async_to_lds_b128 %0, %1, off"
               :
               : "v"(lds_byte_off), "v"((unsigned long long)(size_t)gaddr)
               : "memory");
}
__device__ __forceinline__ void wait_async0() {
  asm volatile("s_wait_asynccnt 0" ::: "memory");
}

// TDM: one descriptor moves a 2D tile (rows x 64 bf16 elems) from global to
// LDS, inserting 4 DWORDs (16B) of padding after every 32 DWORDs (128B) so
// the LDS row pitch is 144B. Wave-scoped (EXEC ignored): caller must branch
// so only one wave issues it. Completion tracked by TENSORcnt.
__device__ __forceinline__ void tdm_load_tile_128x64(unsigned lds_byte_off,
                                                     const void* gaddr,
                                                     unsigned row_stride_elems) {
  unsigned long long ga = (unsigned long long)(size_t)gaddr;
  u32x4 g0;
  g0[0] = 1u;                                   // count=1, user D#, no gather
  g0[1] = lds_byte_off;                         // lds_addr
  g0[2] = (unsigned)ga;                         // global_addr[31:0]
  g0[3] = (unsigned)((ga >> 32) & 0x01FFFFFFu)  // global_addr[56:32]
        | (2u << 30);                           // type = 2 ("image")
  i32x8 g1;
  g1[0] = (1 << 16)      // data_size = 2 bytes
        | (1 << 20)      // pad_enable
        | (4 << 22)      // pad_interval: code 4 = 32 DWORDs (128B)
        | (3 << 25);     // pad_amount:   code 3 = 4 DWORDs (16B)
  // tensor_dim0/1 = 1<<20 (huge: OOB clipping never triggers)
  g1[1] = 0;                                    // abar addr=0, td0[15:0]=0
  g1[2] = 0x10;                                 // td0[31:16]=0x10, td1[15:0]=0
  g1[3] = (64 << 16) | 0x10;                    // td1[31:16]=0x10, tile_dim0=64
  g1[4] = 128;                                  // tile_dim1=128, tile_dim2=0
  g1[5] = (int)row_stride_elems;                // tensor_dim0_stride[31:0]
  g1[6] = 0;                                    // stride[47:32]=0, td1s[15:0]=0
  g1[7] = 0;                                    // td1s[47:16]=0
  asm volatile("tensor_load_to_lds %0, %1" : : "s"(g0), "s"(g1) : "memory");
}
__device__ __forceinline__ void wait_tensor0() {
  __builtin_amdgcn_s_wait_tensorcnt(0);
}

// ---------------------------------------------------------------------------
// f32 -> bf16 conversion (n must be a multiple of 4; all our sizes are)
// ---------------------------------------------------------------------------
__global__ __launch_bounds__(256) void cvt_f32_to_bf16(
    const float* __restrict__ in, __bf16* __restrict__ out, int n) {
  int i = (blockIdx.x * blockDim.x + threadIdx.x) * 4;
  if (i >= n) return;
  float4 f = *(const float4*)(in + i);
  v4bf o;
  o[0] = (__bf16)f.x; o[1] = (__bf16)f.y; o[2] = (__bf16)f.z; o[3] = (__bf16)f.w;
  *(v4bf*)(out + i) = o;
}

// ---------------------------------------------------------------------------
// C[M,N] = A[M,K] * W[N,K]^T   (K = N = 1024, M = 4096), bf16 in, f32 accum.
// 256 threads = 8 waves (4 along M x 2 along N); wave tile 32x64; block tile
// 128x128. K consumed in 64-wide double-buffered stages: A via TDM, W via
// per-lane async copies; both land in 144B-pitch LDS tiles.
//
// MODE 0: outF[m*1024 + n] = acc                      (final projection, f32)
// MODE 1: outB = bf16 [B,H,S,D]  (Q and K head layout)
// MODE 2: outB = bf16 [B,H,D,S]  (V transposed: P*V B-frags become contiguous)
// ---------------------------------------------------------------------------
template <int MODE>
__global__ __launch_bounds__(256) void gemm_xwT(
    const __bf16* __restrict__ A, const __bf16* __restrict__ W,
    float* __restrict__ outF, __bf16* __restrict__ outB) {
  constexpr int Kd = N_EMBD, Nn = N_EMBD;
  constexpr int KSTEP = 64;
  constexpr int RS = KSTEP + 8;                    // 72 elems = 144B row pitch
  __shared__ __align__(16) __bf16 sA[2][128 * RS]; // 2 x 18KB
  __shared__ __align__(16) __bf16 sB[2][128 * RS]; // 2 x 18KB

  int tid  = threadIdx.x;
  int lane = tid & 31;
  int wv   = tid >> 5;                             // 0..7 (wave32)
  int half = lane >> 4, l15 = lane & 15;
  int m0 = (wv & 3) * 32;                          // wave tile inside block
  int n0 = (wv >> 2) * 64;
  int bm = blockIdx.x * 128, bn = blockIdx.y * 128;

  // Stage one 128x64 A tile (TDM, wave 0) + 128x64 W tile (per-lane async).
  // W tile: 1024 16B-chunks / 256 threads = 4 chunks each; 8 chunks/row.
  auto stage = [&](int buf, int k0) {
    if (tid < 32) {                                // wave 0 issues the TDM op
      tdm_load_tile_128x64((unsigned)(size_t)(void*)&sA[buf][0],
                           A + (size_t)bm * Kd + k0, Kd);
    }
    unsigned lb = (unsigned)(size_t)(void*)&sB[buf][0];
#pragma unroll
    for (int i = 0; i < 4; ++i) {
      int cc  = tid + 256 * i;
      int row = cc >> 3, col = cc & 7;             // col in 16B units
      async_ld_b128(lb + row * (RS * 2) + col * 16,
                    W + (size_t)(bn + row) * Kd + k0 + col * 8);
    }
  };

  v8f z = {};
  v8f acc[2][4];
#pragma unroll
  for (int i = 0; i < 2; ++i)
#pragma unroll
    for (int j = 0; j < 4; ++j) acc[i][j] = z;

  stage(0, 0);
  if (tid < 32) wait_tensor0();
  wait_async0();
  __syncthreads();

  for (int ks = 0; ks < Kd / KSTEP; ++ks) {
    if (ks + 1 < Kd / KSTEP) stage((ks + 1) & 1, (ks + 1) * KSTEP);

    const __bf16* a = &sA[ks & 1][0];
    const __bf16* w = &sB[ks & 1][0];
#pragma unroll
    for (int kk = 0; kk < 2; ++kk) {               // two 32-deep WMMA substeps
      v16bf af[2];
#pragma unroll
      for (int mf = 0; mf < 2; ++mf) {
        const __bf16* p = a + (size_t)(m0 + mf * 16 + l15) * RS + kk * 32 + half * 8;
        af[mf] = cat8(*(const v8bf*)p, *(const v8bf*)(p + 16));
      }
      v16bf wf[4];
#pragma unroll
      for (int nf = 0; nf < 4; ++nf) {
        const __bf16* p = w + (size_t)(n0 + nf * 16 + l15) * RS + kk * 32 + half * 8;
        wf[nf] = cat8(*(const v8bf*)p, *(const v8bf*)(p + 16));
      }
#pragma unroll
      for (int mf = 0; mf < 2; ++mf)
#pragma unroll
        for (int nf = 0; nf < 4; ++nf)
          acc[mf][nf] = wmma_bf16(af[mf], wf[nf], acc[mf][nf]);
    }
    if (tid < 32) wait_tensor0();  // drain TDM prefetch (other buffer)
    wait_async0();                 // drain async prefetch (other buffer)
    __syncthreads();               // all waves done reading current buffer
  }

  // Epilogue. C layout: VGPR j -> row (j + 8*half), lane&15 -> col.
#pragma unroll
  for (int mf = 0; mf < 2; ++mf)
#pragma unroll
    for (int nf = 0; nf < 4; ++nf)
#pragma unroll
      for (int j = 0; j < 8; ++j) {
        int m = bm + m0 + mf * 16 + j + 8 * half;
        int n = bn + n0 + nf * 16 + l15;
        float v = acc[mf][nf][j];
        if constexpr (MODE == 0) {
          outF[(size_t)m * Nn + n] = v;
        } else {
          int b = m >> 11, s = m & (SEQ - 1);
          int h = n >> 6,  d = n & (HEAD_DIM - 1);
          size_t off = (MODE == 1)
              ? ((size_t)(b * N_HEAD + h) * SEQ + s) * HEAD_DIM + d
              : ((size_t)(b * N_HEAD + h) * HEAD_DIM + d) * SEQ + s;
          outB[off] = (__bf16)v;
        }
      }
}

// ---------------------------------------------------------------------------
// Flash attention, one wave per 16 query rows. Online softmax in f32.
//   Q,K: bf16 [B,H,S,D]; Vt: bf16 [B,H,D,S]; out: bf16 [B,S,E]
// Per key-tile of 64: 8 WMMA for S = Q*K^T, shfl_xor row reductions,
// P transposed C->A layout via a 2KB LDS tile, 8 WMMA for O += P*V.
// ---------------------------------------------------------------------------
__global__ __launch_bounds__(32) void flash_attn(
    const __bf16* __restrict__ Q, const __bf16* __restrict__ Kmat,
    const __bf16* __restrict__ Vt, __bf16* __restrict__ Out) {
  __shared__ __align__(16) __bf16 sP[16 * 64];
  const int S = SEQ, D = HEAD_DIM;
  int lane = threadIdx.x & 31;
  int half = lane >> 4, l15 = lane & 15;
  int qt = blockIdx.x, h = blockIdx.y, b = blockIdx.z;
  size_t bh = (size_t)(b * N_HEAD + h);
  const __bf16* Qp = Q    + bh * S * D;
  const __bf16* Kp = Kmat + bh * S * D;
  const __bf16* Vp = Vt   + bh * D * S;
  int q0 = qt * 16;

  // Q fragments for the whole kernel (D = 64 -> two 16x32 A-frags)
  v16bf qf[2];
#pragma unroll
  for (int ds = 0; ds < 2; ++ds) {
    const __bf16* p = Qp + (size_t)(q0 + l15) * D + ds * 32 + half * 8;
    qf[ds] = cat8(*(const v8bf*)p, *(const v8bf*)(p + 16));
  }

  v8f z = {};
  v8f o[4];
#pragma unroll
  for (int i = 0; i < 4; ++i) o[i] = z;
  float mrow[8], lrow[8];
#pragma unroll
  for (int j = 0; j < 8; ++j) { mrow[j] = -1e30f; lrow[j] = 0.f; }

  int ktmax = (q0 + 15) >> 6;                 // causal: key tiles 0..ktmax
  for (int kt = 0; kt <= ktmax; ++kt) {
    int kbase = kt * 64;

    // S tile (16 x 64) = Q * K^T : B[k=d][n=kpos] = K[kpos][d] (row-contig)
    v8f sacc[4];
#pragma unroll
    for (int nt = 0; nt < 4; ++nt) {
      v8f c = z;
#pragma unroll
      for (int ds = 0; ds < 2; ++ds) {
        const __bf16* p =
            Kp + (size_t)(kbase + nt * 16 + l15) * D + ds * 32 + half * 8;
        c = wmma_bf16(qf[ds], cat8(*(const v8bf*)p, *(const v8bf*)(p + 16)), c);
      }
      sacc[nt] = c;
    }

    // scale (1/sqrt(64)) + causal mask
#pragma unroll
    for (int nt = 0; nt < 4; ++nt)
#pragma unroll
      for (int j = 0; j < 8; ++j) {
        int kg = kbase + nt * 16 + l15;
        int qg = q0 + j + 8 * half;
        float v = sacc[nt][j] * 0.125f;
        sacc[nt][j] = (kg <= qg) ? v : -1e30f;
      }

    // per-row max across the 16-lane half (row lives across lanes in C layout)
    float rmax[8];
#pragma unroll
    for (int j = 0; j < 8; ++j) {
      float v = fmaxf(fmaxf(sacc[0][j], sacc[1][j]),
                      fmaxf(sacc[2][j], sacc[3][j]));
#pragma unroll
      for (int msk = 1; msk < 16; msk <<= 1)
        v = fmaxf(v, __shfl_xor(v, msk, 32));
      rmax[j] = v;
    }

    // online softmax update
#pragma unroll
    for (int j = 0; j < 8; ++j) {
      float mnew  = fmaxf(mrow[j], rmax[j]);
      float alpha = __expf(mrow[j] - mnew);
      mrow[j] = mnew;
      float rs = 0.f;
#pragma unroll
      for (int nt = 0; nt < 4; ++nt) {
        float e = __expf(sacc[nt][j] - mnew);
        sacc[nt][j] = e;
        rs += e;
      }
#pragma unroll
      for (int msk = 1; msk < 16; msk <<= 1) rs += __shfl_xor(rs, msk, 32);
      lrow[j] = lrow[j] * alpha + rs;
#pragma unroll
      for (int nt = 0; nt < 4; ++nt) o[nt][j] *= alpha;
    }

    // P: C layout -> row-major LDS (bf16). Same-wave DS ops are in-order.
#pragma unroll
    for (int nt = 0; nt < 4; ++nt)
#pragma unroll
      for (int j = 0; j < 8; ++j)
        sP[(j + 8 * half) * 64 + nt * 16 + l15] = (__bf16)sacc[nt][j];

    // reload P as A-fragments (two 16x32 along kpos)
    v16bf pf[2];
#pragma unroll
    for (int ks = 0; ks < 2; ++ks) {
      const __bf16* p = sP + (size_t)l15 * 64 + ks * 32 + half * 8;
      pf[ks] = cat8(*(const v8bf*)p, *(const v8bf*)(p + 16));
    }

    // O += P * V : B[k=kpos][n=dout] = Vt[dout][kpos] (row-contig in Vt)
#pragma unroll
    for (int nt = 0; nt < 4; ++nt)
#pragma unroll
      for (int ks = 0; ks < 2; ++ks) {
        const __bf16* p =
            Vp + (size_t)(nt * 16 + l15) * S + kbase + ks * 32 + half * 8;
        o[nt] = wmma_bf16(pf[ks], cat8(*(const v8bf*)p, *(const v8bf*)(p + 16)),
                          o[nt]);
      }
  }

  // normalize and store as bf16 [B,S,E] for the final projection GEMM
#pragma unroll
  for (int nt = 0; nt < 4; ++nt)
#pragma unroll
    for (int j = 0; j < 8; ++j) {
      int q = q0 + j + 8 * half;
      int e = h * HEAD_DIM + nt * 16 + l15;
      Out[((size_t)b * S + q) * N_EMBD + e] = (__bf16)(o[nt][j] / lrow[j]);
    }
}

// ---------------------------------------------------------------------------
// Host-side orchestration (all launches on `stream`; workspace layout below)
// ---------------------------------------------------------------------------
extern "C" void kernel_launch(void* const* d_in, const int* in_sizes, int n_in,
                              void* d_out, int out_size, void* d_ws,
                              size_t ws_size, hipStream_t stream) {
  (void)in_sizes; (void)n_in; (void)out_size; (void)ws_size;
  const float* x  = (const float*)d_in[0];
  const float* wq = (const float*)d_in[1];
  const float* wk = (const float*)d_in[2];
  const float* wv = (const float*)d_in[3];
  const float* wo = (const float*)d_in[4];
  float* out = (float*)d_out;

  const size_t XE = (size_t)BATCH * SEQ * N_EMBD;   // 4,194,304 elems
  const size_t WE = (size_t)N_EMBD * N_EMBD;        // 1,048,576 elems

  // Workspace (bf16): x | wq wk wv wo | Q | K | Vt | attn  ~= 48 MB total
  __bf16* xb  = (__bf16*)d_ws;
  __bf16* wqb = xb  + XE;
  __bf16* wkb = wqb + WE;
  __bf16* wvb = wkb + WE;
  __bf16* wob = wvb + WE;
  __bf16* Qb  = wob + WE;
  __bf16* Kb  = Qb  + XE;
  __bf16* Vtb = Kb  + XE;
  __bf16* Ab  = Vtb + XE;

  // 1) down-convert to bf16 (everything stays L2-resident afterwards)
  cvt_f32_to_bf16<<<(XE / 4 + 255) / 256, 256, 0, stream>>>(x,  xb,  (int)XE);
  cvt_f32_to_bf16<<<(WE / 4 + 255) / 256, 256, 0, stream>>>(wq, wqb, (int)WE);
  cvt_f32_to_bf16<<<(WE / 4 + 255) / 256, 256, 0, stream>>>(wk, wkb, (int)WE);
  cvt_f32_to_bf16<<<(WE / 4 + 255) / 256, 256, 0, stream>>>(wv, wvb, (int)WE);
  cvt_f32_to_bf16<<<(WE / 4 + 255) / 256, 256, 0, stream>>>(wo, wob, (int)WE);

  // 2) QKV projections (TDM+async WMMA GEMMs); V written transposed [B,H,D,S]
  dim3 gg(MROWS / 128, N_EMBD / 128);   // (32, 8)
  gemm_xwT<1><<<gg, 256, 0, stream>>>(xb, wqb, nullptr, Qb);
  gemm_xwT<1><<<gg, 256, 0, stream>>>(xb, wkb, nullptr, Kb);
  gemm_xwT<2><<<gg, 256, 0, stream>>>(xb, wvb, nullptr, Vtb);

  // 3) causal flash attention (one wave per 16 query rows)
  flash_attn<<<dim3(SEQ / 16, N_HEAD, BATCH), 32, 0, stream>>>(Qb, Kb, Vtb, Ab);

  // 4) output projection -> f32 d_out
  gemm_xwT<0><<<gg, 256, 0, stream>>>(Ab, wob, out, nullptr);
}
